// Net_78108275245565
// MI455X (gfx1250) — matrix operand order repaired and checked by reading, not compiled, not run on hardware
//
#include <hip/hip_runtime.h>
#include <hip/hip_bf16.h>
#include <math.h>

#define DIMV 70
#define NE 6
#define BATCH 8
#define TARGETN 29

typedef __attribute__((ext_vector_type(16))) __bf16 v16bf;
typedef __attribute__((ext_vector_type(8)))  float  v8f;

__device__ __forceinline__ unsigned short f32_to_bf16_raw(float f) {
  unsigned int u = __float_as_uint(f);
  unsigned int r = u + 0x7FFFu + ((u >> 16) & 1u);   // round-to-nearest-even
  return (unsigned short)(r >> 16);
}
__device__ __forceinline__ float bf16_raw_to_f32(unsigned short h) {
  return __uint_as_float(((unsigned int)h) << 16);
}

// ---------------- Gaussian table: G[e][a][i] = exp(-((a-35)-i)^2 / (2*sigma_e^2))
__global__ void g_kernel(const float* __restrict__ sigma, float* __restrict__ G) {
  int id = blockIdx.x * blockDim.x + threadIdx.x;
  if (id >= NE * DIMV * DIMV) return;
  int i = id % DIMV;
  int a = (id / DIMV) % DIMV;
  int e = id / (DIMV * DIMV);
  float d = (float)(a - DIMV / 2 - i);
  float s = sigma[e];
  G[id] = expf(-(d * d) / (2.0f * s * s));
}

// ---------------- blur pass 1: contract z. out[b,e,a,j,k] = sum_i G[e,a,i]*x[b,e,i,j,k]
__global__ void blur_z(const float* __restrict__ x, const float* __restrict__ G,
                       float* __restrict__ out) {
  int id = blockIdx.x * blockDim.x + threadIdx.x;
  if (id >= BATCH * NE * DIMV * DIMV * DIMV) return;
  int k = id % DIMV;
  int j = (id / DIMV) % DIMV;
  int a = (id / (DIMV * DIMV)) % DIMV;
  int e = (id / (DIMV * DIMV * DIMV)) % NE;
  int b = id / (NE * DIMV * DIMV * DIMV);
  const float* g  = G + (e * DIMV + a) * DIMV;
  const float* xp = x + (b * NE + e) * (DIMV * DIMV * DIMV) + j * DIMV + k;
  float acc = 0.f;
  for (int i = 0; i < DIMV; ++i) acc = fmaf(g[i], xp[i * DIMV * DIMV], acc);
  out[id] = acc;
}

// ---------------- blur pass 2: contract y. out[b,e,a,p,k] = sum_j G[e,p,j]*t[b,e,a,j,k]
__global__ void blur_y(const float* __restrict__ t, const float* __restrict__ G,
                       float* __restrict__ out) {
  int id = blockIdx.x * blockDim.x + threadIdx.x;
  if (id >= BATCH * NE * DIMV * DIMV * DIMV) return;
  int k = id % DIMV;
  int p = (id / DIMV) % DIMV;
  int a = (id / (DIMV * DIMV)) % DIMV;
  int e = (id / (DIMV * DIMV * DIMV)) % NE;
  int b = id / (NE * DIMV * DIMV * DIMV);
  const float* g  = G + (e * DIMV + p) * DIMV;
  const float* tp = t + (b * NE + e) * (DIMV * DIMV * DIMV) + a * (DIMV * DIMV) + k;
  float acc = 0.f;
  for (int j = 0; j < DIMV; ++j) acc = fmaf(g[j], tp[j * DIMV], acc);
  out[id] = acc;
}

// ---------------- blur pass 3: contract x, transpose to channel-last (PADDED to 8 ch), bf16.
// cube[b][a][p][q][e8] (bf16 raw), e8 in 0..7, channels 6,7 are zero pad.
__global__ void blur_x(const float* __restrict__ t, const float* __restrict__ G,
                       unsigned short* __restrict__ cube) {
  int id = blockIdx.x * blockDim.x + threadIdx.x;
  if (id >= BATCH * DIMV * DIMV * DIMV * 8) return;
  int e = id & 7;
  int q = (id >> 3) % DIMV;
  int p = (id / (8 * DIMV)) % DIMV;
  int a = (id / (8 * DIMV * DIMV)) % DIMV;
  int b = id / (8 * DIMV * DIMV * DIMV);
  unsigned short o = 0;
  if (e < NE) {
    const float* g  = G + (e * DIMV + q) * DIMV;
    const float* tp = t + (b * NE + e) * (DIMV * DIMV * DIMV) + a * (DIMV * DIMV) + p * DIMV;
    float acc = 0.f;
    for (int k = 0; k < DIMV; ++k) acc = fmaf(g[k], tp[k], acc);
    o = f32_to_bf16_raw(acc);
  }
  cube[id] = o;
}

// ---------------- weight transform: w[Cout][Cin][3][3][3] f32 -> wm2[chunk][n][32] bf16
// wm2[((k/32)*COUT + n)*32 + (k%32)] = (t<27 && c<CIN) ? w[n][c][t] : 0,  k = t*CINP + c.
// Zero rows kill contributions of padded K, so the WMMA kernel needs NO guards.
template<int CIN, int CINP, int COUT>
__global__ void wt_kernel(const float* __restrict__ w, unsigned short* __restrict__ wm2) {
  constexpr int KPAD = ((CINP * 27 + 31) / 32) * 32;
  constexpr int NCH = KPAD / 32;
  int id = blockIdx.x * blockDim.x + threadIdx.x;
  if (id >= NCH * COUT * 32) return;
  int lk = id & 31;
  int n  = (id >> 5) % COUT;
  int ch = id / (32 * COUT);
  int k  = ch * 32 + lk;
  int t  = k / CINP;            // CINP is a power of two
  int c  = k % CINP;
  float v = 0.f;
  if (t < 27 && c < CIN) v = w[(n * CIN + c) * 27 + t];
  wm2[id] = f32_to_bf16_raw(v);
}

// ---------------- fused conv3x3x3 (VALID) + bias + 2x2x2 maxpool + ReLU, bf16 WMMA.
// Input : channel-last bf16 [B][SIN][SIN][SIN][CINP] (CINP = padded channels, mult of 8)
// Weights: wm2[chunk][n][32] bf16 (zero-padded K rows)
// Output : channel-last bf16 [B][P][P][P][COUT]
// Block = (b, pz, py, pooled-x-pair); waves = COUT/16.
// A fragment (16-bit 16x32 layout): per lane = TWO contiguous 8-channel runs -> 2x ds_load_b128.
// B fragment: per lane = TWO 16-byte runs of wm2 (K = khalf*8 and 16 + khalf*8) -> 2x global_load_b128.
// Patch staged via global_load_async_to_lds_b128 (ASYNCcnt) with zero-fill for OOB vectors.
template<int CINP, int COUT, int SIN, int P>
__global__ void __launch_bounds__(32 * (COUT / 16))
conv_pool_kernel(const unsigned short* __restrict__ in,
                 const unsigned short* __restrict__ wm2,
                 const float* __restrict__ bias,
                 unsigned short* __restrict__ out) {
  constexpr int KPAD = ((CINP * 27 + 31) / 32) * 32;
  constexpr int NCH  = KPAD / 32;
  constexpr int PXP  = (P + 1) / 2;
  constexpr int NW   = COUT / 16;
  constexpr int SHIFT = (CINP == 8) ? 3 : (CINP == 32) ? 5 : (CINP == 64) ? 6 : 7;
  constexpr int NV   = 4 * 4 * 6 * (CINP / 8);   // 16-byte vectors in the patch

  __shared__ __align__(16) unsigned short patch[4 * 4 * 6 * CINP];

  int bid = blockIdx.x;
  int pxp = bid % PXP; bid /= PXP;
  int py  = bid % P;   bid /= P;
  int pz  = bid % P;   bid /= P;
  int b   = bid;

  int tid  = (int)threadIdx.x;
  int lane = tid & 31;
  int wave = tid >> 5;

  // ---- async patch stage: input z in [2pz,2pz+3], y likewise, x in [4pxp,4pxp+5]
  int z0 = 2 * pz, y0 = 2 * py, x0 = 4 * pxp;
  for (int v = tid; v < NV; v += 32 * NW) {
    int c8 = v & (CINP / 8 - 1);
    int r  = v / (CINP / 8);
    int xx = r % 6; r /= 6;
    int yy = r & 3;
    int zz = r >> 2;
    int gz = z0 + zz, gy = y0 + yy, gx = x0 + xx;
    int lidx = ((zz * 4 + yy) * 6 + xx) * CINP + c8 * 8;
    if (gz < SIN && gy < SIN && gx < SIN) {
      const unsigned short* src =
          in + ((((size_t)b * SIN + gz) * SIN + gy) * SIN + gx) * CINP + c8 * 8;
      unsigned int       loff = (unsigned int)(uintptr_t)(patch + lidx);
      unsigned long long gadr = (unsigned long long)(uintptr_t)src;
      asm volatile("global_load_async_to_lds_b128 %0, %1, off"
                   :: "v"(loff), "v"(gadr) : "memory");
    } else {
      uint4 z4 = {0u, 0u, 0u, 0u};
      *(uint4*)(patch + lidx) = z4;
    }
  }
  asm volatile("s_wait_asynccnt 0" ::: "memory");
  __syncthreads();

  // M decode: m = dz*8 + dy*4 + dx (2x2 z/y, 4 conv-x positions = 2 pooled x)
  int m  = lane & 15;
  int dz = m >> 3, dy = (m >> 2) & 1, dx = m & 3;
  int khalf = lane >> 4;
  int nbase = wave * 16 + (lane & 15);

  v8f acc = {0.f, 0.f, 0.f, 0.f, 0.f, 0.f, 0.f, 0.f};
  const unsigned short* wbase = wm2 + (size_t)nbase * 32 + (size_t)khalf * 8;

  for (int ch = 0; ch < NCH; ++ch) {
    int k0 = ch * 32;
    union { uint4 q[2]; v16bf v; } af, bfv;

    // A: run0 covers elements 0..7 (K = k0 + khalf*8 + 0..7),
    //    run1 covers elements 8..15 (K = k0 + 16 + khalf*8 + 0..7).
    {
      int k  = k0 + khalf * 8;
      int t  = k >> SHIFT; t = (t > 26) ? 26 : t;      // branch-free clamp; padded K has B==0
      int c0 = k & (CINP - 1);
      int tz = t / 9, tr = t - tz * 9, ty = tr / 3, tx = tr - ty * 3;
      af.q[0] = *(const uint4*)(patch + (((dz + tz) * 4 + (dy + ty)) * 6 + (dx + tx)) * CINP + c0);

      k  = k0 + 16 + khalf * 8;
      t  = k >> SHIFT; t = (t > 26) ? 26 : t;
      c0 = k & (CINP - 1);
      tz = t / 9; tr = t - tz * 9; ty = tr / 3; tx = tr - ty * 3;
      af.q[1] = *(const uint4*)(patch + (((dz + tz) * 4 + (dy + ty)) * 6 + (dx + tx)) * CINP + c0);
    }

    // B: two 16-byte runs (K = khalf*8 + 0..7, 16 + khalf*8 + 0..7) for this (chunk, n)
    const uint4* wp = (const uint4*)(wbase + (size_t)ch * COUT * 32);
    bfv.q[0] = wp[0];
    bfv.q[1] = wp[2];

    int chn = (ch + 1 < NCH) ? (ch + 1) : ch;          // branch-free next-chunk prefetch
    __builtin_prefetch(wbase + (size_t)chn * COUT * 32, 0, 1);

    acc = __builtin_amdgcn_wmma_f32_16x16x32_bf16(false, af.v, false, bfv.v,
                                                  (short)0, acc, false, false);
  }

  // C layout: lane holds m = r + 8*khalf, n = lane&15.
  // pooled sel by conv-x pair: r in {0,1,4,5} -> pooled x0; {2,3,6,7} -> pooled x0+1.
  float p0 = fmaxf(fmaxf(acc[0], acc[1]), fmaxf(acc[4], acc[5]));
  float p1 = fmaxf(fmaxf(acc[2], acc[3]), fmaxf(acc[6], acc[7]));
  p0 = fmaxf(p0, __shfl_xor(p0, 16, 32));  // fold m(0..7) with m(8..15) halves (wave32)
  p1 = fmaxf(p1, __shfl_xor(p1, 16, 32));

  if (lane < 16) {
    int n = wave * 16 + lane;
    float bv = bias[n];
    p0 = fmaxf(p0 + bv, 0.f);
    p1 = fmaxf(p1 + bv, 0.f);
    int px0 = 2 * pxp;
    out[(((size_t)(b * P + pz) * P + py) * P + px0) * COUT + n] = f32_to_bf16_raw(p0);
    if (px0 + 1 < P)
      out[(((size_t)(b * P + pz) * P + py) * P + px0 + 1) * COUT + n] = f32_to_bf16_raw(p1);
  }
}

// ---------------- FC1: y1[b][j] = relu( sum_v v[b][v]*W[j][v] + b[j] )
// v = c*8 + (z*4+y*2+x) mapped onto channel-last h4[b][z][y][x][c].
__global__ void fc1_kernel(const unsigned short* __restrict__ h4,
                           const float* __restrict__ w, const float* __restrict__ bias,
                           float* __restrict__ y1) {
  int id = blockIdx.x * blockDim.x + threadIdx.x;
  if (id >= BATCH * 1024) return;
  int j = id % 1024;
  int b = id / 1024;
  float acc = 0.f;
  for (int v = 0; v < 2048; ++v) {
    int c = v >> 3;
    int s = v & 7;
    acc = fmaf(bf16_raw_to_f32(h4[(b * 8 + s) * 256 + c]), w[j * 2048 + v], acc);
  }
  y1[id] = fmaxf(acc + bias[j], 0.f);
}

// ---------------- FC2: out[b][o] = sum_j y1[b][j]*W2[o][j] + b2[o]
__global__ void fc2_kernel(const float* __restrict__ y1, const float* __restrict__ w,
                           const float* __restrict__ bias, float* __restrict__ outp) {
  int id = blockIdx.x * blockDim.x + threadIdx.x;
  if (id >= BATCH * TARGETN) return;
  int o = id % TARGETN;
  int b = id / TARGETN;
  float acc = 0.f;
  for (int j = 0; j < 1024; ++j)
    acc = fmaf(y1[b * 1024 + j], w[o * 1024 + j], acc);
  outp[id] = acc + bias[o];
}

extern "C" void kernel_launch(void* const* d_in, const int* in_sizes, int n_in,
                              void* d_out, int out_size, void* d_ws, size_t ws_size,
                              hipStream_t stream) {
  (void)in_sizes; (void)n_in; (void)out_size; (void)ws_size;

  const float* x     = (const float*)d_in[0];
  const float* sigma = (const float*)d_in[1];
  const float* c1w   = (const float*)d_in[2];
  const float* c1b   = (const float*)d_in[3];
  const float* c2w   = (const float*)d_in[4];
  const float* c2b   = (const float*)d_in[5];
  const float* c3w   = (const float*)d_in[6];
  const float* c3b   = (const float*)d_in[7];
  const float* c4w   = (const float*)d_in[8];
  const float* c4b   = (const float*)d_in[9];
  const float* f1w   = (const float*)d_in[10];
  const float* f1b   = (const float*)d_in[11];
  const float* f2w   = (const float*)d_in[12];
  const float* f2b   = (const float*)d_in[13];

  char* ws = (char*)d_ws;
  size_t off = 0;
  auto carve = [&](size_t bytes) -> char* {
    char* p = ws + off;
    off += (bytes + 255) & ~(size_t)255;
    return p;
  };

  const size_t NVOX = (size_t)BATCH * NE * DIMV * DIMV * DIMV;  // 16,464,000

  float*          G    = (float*)carve((size_t)NE * DIMV * DIMV * 4);
  unsigned short* w1m  = (unsigned short*)carve((size_t)7   * 32  * 32 * 2);
  unsigned short* w2m  = (unsigned short*)carve((size_t)27  * 64  * 32 * 2);
  unsigned short* w3m  = (unsigned short*)carve((size_t)54  * 128 * 32 * 2);
  unsigned short* w4m  = (unsigned short*)carve((size_t)108 * 256 * 32 * 2);
  float*          y1   = (float*)carve((size_t)BATCH * 1024 * 4);
  float*          bufA = (float*)carve(NVOX * 4);
  float*          bufB = (float*)carve(NVOX * 4);

  // Buffer reuse (each target is dead before being overwritten):
  unsigned short* cube = (unsigned short*)bufA;  // [8][70][70][70][8] bf16 (43.9 MB <= 65.8 MB)
  unsigned short* h1   = (unsigned short*)bufB;  // [8][34][34][34][32]
  unsigned short* h2   = (unsigned short*)bufA;  // [8][16][16][16][64]
  unsigned short* h3   = (unsigned short*)bufB;  // [8][7][7][7][128]
  unsigned short* h4   = (unsigned short*)bufA;  // [8][2][2][2][256]

  const int T = 256;
  g_kernel<<<(NE * DIMV * DIMV + T - 1) / T, T, 0, stream>>>(sigma, G);

  int nb = (int)((NVOX + T - 1) / T);
  blur_z<<<nb, T, 0, stream>>>(x,    G, bufA);
  blur_y<<<nb, T, 0, stream>>>(bufA, G, bufB);
  int nbc = (int)(((size_t)BATCH * DIMV * DIMV * DIMV * 8 + T - 1) / T);
  blur_x<<<nbc, T, 0, stream>>>(bufB, G, cube);

  wt_kernel<6, 8, 32>    <<<(7   * 32  * 32 + T - 1) / T, T, 0, stream>>>(c1w, w1m);
  wt_kernel<32, 32, 64>  <<<(27  * 64  * 32 + T - 1) / T, T, 0, stream>>>(c2w, w2m);
  wt_kernel<64, 64, 128> <<<(54  * 128 * 32 + T - 1) / T, T, 0, stream>>>(c3w, w3m);
  wt_kernel<128, 128, 256><<<(108 * 256 * 32 + T - 1) / T, T, 0, stream>>>(c4w, w4m);

  conv_pool_kernel<8, 32, 70, 34>  <<<BATCH * 34 * 34 * 17, 64,  0, stream>>>(cube, w1m, c1b, h1);
  conv_pool_kernel<32, 64, 34, 16> <<<BATCH * 16 * 16 * 8,  128, 0, stream>>>(h1,   w2m, c2b, h2);
  conv_pool_kernel<64, 128, 16, 7> <<<BATCH * 7 * 7 * 4,    256, 0, stream>>>(h2,   w3m, c3b, h3);
  conv_pool_kernel<128, 256, 7, 2> <<<BATCH * 2 * 2 * 1,    512, 0, stream>>>(h3,   w4m, c4b, h4);

  fc1_kernel<<<(BATCH * 1024 + T - 1) / T, T, 0, stream>>>(h4, f1w, f1b, y1);
  fc2_kernel<<<1, T, 0, stream>>>(y1, f2w, f2b, (float*)d_out);
}
